// MambaBlock_58059367907561
// MI455X (gfx1250) — compile-verified, matrix operand order checked by reading, not run on hardware
//
#include <hip/hip_runtime.h>
#include <hip/hip_bf16.h>
#include <math.h>

// ---------------------------------------------------------------------------
// Mamba block forward for MI455X (gfx1250, wave32, WMMA, bf16 data path).
// B=4, L=2048, DIM=1024, D_INNER=2048, D_STATE=16, DT_RANK=128.
// ---------------------------------------------------------------------------

#define MB_B      4
#define MB_L      2048
#define MB_DIM    1024
#define MB_DI     2048
#define MB_DS     16
#define MB_DTR    128
#define MB_NXP    160
#define MB_M      (MB_B * MB_L)   // 8192

typedef __attribute__((ext_vector_type(16))) __bf16 v16bf;
typedef __attribute__((ext_vector_type(8)))  float  v8f;
typedef __attribute__((ext_vector_type(4)))  unsigned int uint4v;
typedef __attribute__((ext_vector_type(2)))  unsigned int uint2v;

struct U32x8 { uint4v lo, hi; };

__device__ __forceinline__ unsigned short f2bf(float f) {
    unsigned int u = __builtin_bit_cast(unsigned int, f);
    unsigned int r = u + 0x7FFFu + ((u >> 16) & 1u);   // RNE
    return (unsigned short)(r >> 16);
}

// ---------------------------------------------------------------------------
// Vectorized f32 -> bf16 convert (8 elements / thread, b128 store).
// ---------------------------------------------------------------------------
__global__ __launch_bounds__(256)
void cvt_f32_bf16(const float* __restrict__ src, unsigned short* __restrict__ dst,
                  int n8)
{
    const int i = blockIdx.x * 256 + threadIdx.x;
    if (i >= n8) return;
    const float4* s = (const float4*)src + (size_t)i * 2;
    const float4 a = s[0], b = s[1];
    uint4v o;
    o[0] = (unsigned int)f2bf(a.x) | ((unsigned int)f2bf(a.y) << 16);
    o[1] = (unsigned int)f2bf(a.z) | ((unsigned int)f2bf(a.w) << 16);
    o[2] = (unsigned int)f2bf(b.x) | ((unsigned int)f2bf(b.y) << 16);
    o[3] = (unsigned int)f2bf(b.z) | ((unsigned int)f2bf(b.w) << 16);
    *((uint4v*)dst + i) = o;
}

// ---------------------------------------------------------------------------
// WMMA GEMM, bf16 A[M,K] (row stride LDA) x bf16 B[K,N] + bias -> f32 C[M,N].
// Compile-time shape (N, K, LDA). ACT: 0=identity, 1=softplus.
// NBCHK: column bounds check (only N=160 GEMM). NBF>0: also write bf16 copy
// of the first NBF columns to Cb. M % 128 == 0 assumed.
// Block tile 128x64, BK=32, 256 threads = 8 waves, wave tile 32x32 (2x2 WMMA).
// Software-pipelined: next K-tile global loads overlap fragment/WMMA phase.
// ---------------------------------------------------------------------------
template <int ACT, bool NBCHK, int NBF, int N, int K, int LDA>
__global__ __launch_bounds__(256)
void mamba_wmma_gemm(const unsigned short* __restrict__ A,
                     const unsigned short* __restrict__ B,
                     const float* __restrict__ bias, float* __restrict__ C,
                     unsigned short* __restrict__ Cb)
{
    constexpr int BK = 32;
    constexpr int SAS = 40, SBS = 40;               // padded LDS strides (shorts)
    __shared__ __attribute__((aligned(16))) unsigned short sA[128 * SAS];
    __shared__ __attribute__((aligned(16))) unsigned short sB[64 * SBS];

    const int tid  = threadIdx.x;
    const int lane = tid & 31;
    const int w    = tid >> 5;
    const int bm   = blockIdx.y * 128;
    const int bn   = blockIdx.x * 64;
    const int wm   = (w >> 1) * 32;
    const int wn   = (w & 1)  * 32;

    v8f acc[2][2] = {};

    // staging assignments
    const int arow = tid >> 1;                      // 0..127
    const int ah   = (tid & 1) * 16;                // K half of A row
    const int k2   = (tid >> 4) * 2;                // 0,2,..,30
    const int n0   = (tid & 15) * 4;                // 0,4,..,60

    const int koff = (lane < 16) ? 0 : 8;
    const int mrow = lane & 15;

    const unsigned short* aSrc = A + (size_t)(bm + arow) * LDA + ah;
    const unsigned short* bSrc = B + (size_t)k2 * N + bn + n0;
    const bool bInb = !NBCHK || (bn + n0) < N;

    uint4v a0, a1;            // prefetched A regs
    uint2v b0 = {0u, 0u}, b1 = {0u, 0u};
    a0 = *(const uint4v*)(aSrc);
    a1 = *(const uint4v*)(aSrc + 8);
    if (bInb) {
        b0 = *(const uint2v*)(bSrc);
        b1 = *(const uint2v*)(bSrc + N);
    }

    for (int kt = 0; kt < K; kt += BK) {
        // ---- store prefetched tile to LDS ----
        {
            unsigned short* dst = &sA[arow * SAS + ah];
            *(uint4v*)(dst)     = a0;
            *(uint4v*)(dst + 8) = a1;
        }
        #pragma unroll
        for (int j = 0; j < 2; ++j) {
            // pair (k2, k2+1) for n = n0+2j, n0+2j+1 via v_perm_b32
            const unsigned int pe = __builtin_amdgcn_perm(b1[j], b0[j], 0x05040100u);
            const unsigned int po = __builtin_amdgcn_perm(b1[j], b0[j], 0x07060302u);
            *(unsigned int*)&sB[(n0 + 2 * j + 0) * SBS + k2] = pe;
            *(unsigned int*)&sB[(n0 + 2 * j + 1) * SBS + k2] = po;
        }
        __syncthreads();

        // ---- prefetch next tile (overlaps fragment loads + WMMA) ----
        if (kt + BK < K) {
            a0 = *(const uint4v*)(aSrc + kt + BK);
            a1 = *(const uint4v*)(aSrc + kt + BK + 8);
            if (bInb) {
                const unsigned short* p = bSrc + (size_t)(kt + BK) * N;
                b0 = *(const uint2v*)(p);
                b1 = *(const uint2v*)(p + N);
            }
        }

        // ---- fragments: two b128 LDS reads each ----
        v16bf aF[2], bF[2];
        #pragma unroll
        for (int mi = 0; mi < 2; ++mi) {
            const unsigned short* p = &sA[(wm + mi * 16 + mrow) * SAS + koff];
            U32x8 r;
            r.lo = *(const uint4v*)(p);
            r.hi = *(const uint4v*)(p + 16);
            aF[mi] = __builtin_bit_cast(v16bf, r);
        }
        #pragma unroll
        for (int ni = 0; ni < 2; ++ni) {
            const unsigned short* p = &sB[(wn + ni * 16 + mrow) * SBS + koff];
            U32x8 r;
            r.lo = *(const uint4v*)(p);
            r.hi = *(const uint4v*)(p + 16);
            bF[ni] = __builtin_bit_cast(v16bf, r);
        }

        #pragma unroll
        for (int mi = 0; mi < 2; ++mi)
            #pragma unroll
            for (int ni = 0; ni < 2; ++ni)
                acc[mi][ni] = __builtin_amdgcn_wmma_f32_16x16x32_bf16(
                    false, aF[mi], false, bF[ni], (short)0, acc[mi][ni],
                    false, false);
        __syncthreads();
    }

    // ---- epilogue: VGPR r -> M = r + 8*(lane>=16), N = lane&15 ----
    const int ncol = lane & 15;
    const int rofs = (lane >> 4) * 8;
    #pragma unroll
    for (int mi = 0; mi < 2; ++mi) {
        #pragma unroll
        for (int ni = 0; ni < 2; ++ni) {
            const int col = bn + wn + ni * 16 + ncol;
            if (!NBCHK || col < N) {
                const float bvv = bias[col];
                const int row0 = bm + wm + mi * 16 + rofs;
                float* cp = C + (size_t)row0 * N + col;
                unsigned short* cbp =
                    (NBF > 0) ? Cb + (size_t)row0 * NBF + col : nullptr;
                #pragma unroll
                for (int r = 0; r < 8; ++r) {
                    float v = acc[mi][ni][r] + bvv;
                    if (ACT == 1)
                        v = (v > 20.f) ? v : log1pf(__expf(v));
                    *cp = v;
                    cp += N;
                    if (NBF > 0) {
                        if (col < NBF) *cbp = f2bf(v);
                        cbp += NBF;
                    }
                }
            }
        }
    }
}

// ---------------------------------------------------------------------------
// Depthwise causal conv1d (width 4) + SiLU. Writes f32 xa and bf16 xab.
// ---------------------------------------------------------------------------
__global__ __launch_bounds__(256)
void mamba_conv_silu(const float* __restrict__ xz, const float* __restrict__ ck,
                     const float* __restrict__ cb, float* __restrict__ xa,
                     unsigned short* __restrict__ xab)
{
    const int i = blockIdx.x * 256 + threadIdx.x;        // over B*L*DI
    const int c  = i & (MB_DI - 1);
    const int bl = i >> 11;
    const int l  = bl & (MB_L - 1);

    float acc = cb[c];
    #pragma unroll
    for (int w = 0; w < 4; ++w) {
        const int ll = l + w - 3;
        if (ll >= 0)
            acc += ck[w * MB_DI + c] *
                   xz[(size_t)(bl + w - 3) * (2 * MB_DI) + c];
    }
    const float s = acc / (1.f + __expf(-acc));          // SiLU
    xa[(size_t)i]  = s;
    xab[(size_t)i] = f2bf(s);
}

// ---------------------------------------------------------------------------
// Selective scan: one thread per (b, d_inner) channel; 16 states in registers.
// ---------------------------------------------------------------------------
__global__ __launch_bounds__(256)
void mamba_scan(const float* __restrict__ xa, const float* __restrict__ dt,
                const float* __restrict__ dtBC, const float* __restrict__ A_log,
                float* __restrict__ y)
{
    const int g  = blockIdx.x * 256 + threadIdx.x;       // 0..B*DI-1
    const int b  = g >> 11;
    const int di = g & (MB_DI - 1);

    float Arow[MB_DS];
    #pragma unroll
    for (int s = 0; s < MB_DS; ++s)
        Arow[s] = -__expf(A_log[di * MB_DS + s]);

    float h[MB_DS];
    #pragma unroll
    for (int s = 0; s < MB_DS; ++s) h[s] = 0.f;

    for (int l = 0; l < MB_L; ++l) {
        const size_t idx = (size_t)b * MB_L + l;
        const float dt_t = dt[idx * MB_DI + di];
        const float x_t  = xa[idx * MB_DI + di];
        if (l + 8 < MB_L) {                               // global_prefetch_b8
            __builtin_prefetch(dt + (idx + 8) * MB_DI + di, 0, 0);
            __builtin_prefetch(xa + (idx + 8) * MB_DI + di, 0, 0);
        }
        const float4* pb4 = (const float4*)(dtBC + idx * MB_NXP + MB_DTR);
        float Bv[MB_DS], Cv[MB_DS];
        *(float4*)&Bv[0]  = pb4[0]; *(float4*)&Bv[4]  = pb4[1];
        *(float4*)&Bv[8]  = pb4[2]; *(float4*)&Bv[12] = pb4[3];
        *(float4*)&Cv[0]  = pb4[4]; *(float4*)&Cv[4]  = pb4[5];
        *(float4*)&Cv[8]  = pb4[6]; *(float4*)&Cv[12] = pb4[7];

        const float dtx = dt_t * x_t;
        float ysum = 0.f;
        #pragma unroll
        for (int s = 0; s < MB_DS; ++s) {
            const float hb = __expf(dt_t * Arow[s]) * h[s] + dtx * Bv[s];
            h[s] = hb;
            ysum += Cv[s] * hb;
        }
        y[idx * MB_DI + di] = ysum;
    }
}

// ---------------------------------------------------------------------------
// Gate: ygb = bf16( (y + xa*D) * silu(z) ),  z = xz[:, :, 2048:4096].
// ---------------------------------------------------------------------------
__global__ __launch_bounds__(256)
void mamba_gate(const float* __restrict__ xz, const float* __restrict__ xa,
                const float* __restrict__ D_ssm, const float* __restrict__ y,
                unsigned short* __restrict__ ygb)
{
    const int i = blockIdx.x * 256 + threadIdx.x;
    const int c   = i & (MB_DI - 1);
    const int row = i >> 11;
    const float z = xz[(size_t)row * (2 * MB_DI) + MB_DI + c];
    const float g = z / (1.f + __expf(-z));
    ygb[(size_t)i] = f2bf((y[(size_t)i] + xa[(size_t)i] * D_ssm[c]) * g);
}

// ---------------------------------------------------------------------------
extern "C" void kernel_launch(void* const* d_in, const int* in_sizes, int n_in,
                              void* d_out, int out_size, void* d_ws, size_t ws_size,
                              hipStream_t stream)
{
    const float* x      = (const float*)d_in[0];
    const float* W_in   = (const float*)d_in[1];
    const float* b_in   = (const float*)d_in[2];
    const float* conv_k = (const float*)d_in[3];
    const float* conv_b = (const float*)d_in[4];
    const float* A_log  = (const float*)d_in[5];
    const float* W_xp   = (const float*)d_in[6];
    const float* b_xp   = (const float*)d_in[7];
    const float* W_dt   = (const float*)d_in[8];
    const float* b_dt   = (const float*)d_in[9];
    const float* D_ssm  = (const float*)d_in[10];
    const float* W_out  = (const float*)d_in[11];
    const float* b_out  = (const float*)d_in[12];
    float* out = (float*)d_out;

    // -------- workspace layout --------
    float* ws   = (float*)d_ws;
    float* xz   = ws;                                    // [8192,4096] f32
    float* xa   = xz   + (size_t)MB_M * 2 * MB_DI;       // [8192,2048] f32
    float* dtBC = xa   + (size_t)MB_M * MB_DI;           // [8192,160]  f32
    float* dt   = dtBC + (size_t)MB_M * MB_NXP;          // [8192,2048] f32
    float* y    = dt   + (size_t)MB_M * MB_DI;           // [8192,2048] f32
    unsigned short* wb_in  = (unsigned short*)(y + (size_t)MB_M * MB_DI);
    unsigned short* wb_xp  = wb_in  + (size_t)MB_DIM * 2 * MB_DI;
    unsigned short* wb_dt  = wb_xp  + (size_t)MB_DI * MB_NXP;
    unsigned short* wb_out = wb_dt  + (size_t)MB_DTR * MB_DI;
    unsigned short* xb     = wb_out + (size_t)MB_DI * MB_DIM;   // x as bf16
    unsigned short* xab    = xb     + (size_t)MB_M * MB_DIM;    // silu(conv)
    unsigned short* dtib   = xab    + (size_t)MB_M * MB_DI;     // dt_int bf16
    unsigned short* ygb    = dtib   + (size_t)MB_M * MB_DTR;    // gated y bf16

    const dim3 blk(256);
    auto cvt = [&](const float* s, unsigned short* d, size_t n) {
        const int n8 = (int)(n / 8);
        cvt_f32_bf16<<<(n8 + 255) / 256, blk, 0, stream>>>(s, d, n8);
    };

    // 0) weight / input conversions to bf16
    cvt(W_in,  wb_in,  (size_t)MB_DIM * 2 * MB_DI);
    cvt(W_xp,  wb_xp,  (size_t)MB_DI * MB_NXP);
    cvt(W_dt,  wb_dt,  (size_t)MB_DTR * MB_DI);
    cvt(W_out, wb_out, (size_t)MB_DI * MB_DIM);
    cvt(x,     xb,     (size_t)MB_M * MB_DIM);

    // 1) xz = x @ W_in + b_in           (8192 x 4096 x 1024)
    mamba_wmma_gemm<0, false, 0, 2 * MB_DI, MB_DIM, MB_DIM>
        <<<dim3((2 * MB_DI) / 64, MB_M / 128), blk, 0, stream>>>(
            xb, wb_in, b_in, xz, nullptr);

    // 2) xa/xab = silu(causal_dwconv(xz[:, :, :2048]))
    mamba_conv_silu<<<(MB_M * MB_DI) / 256, blk, 0, stream>>>(xz, conv_k, conv_b, xa, xab);

    // 3) dtBC = xa @ W_xp + b_xp        (8192 x 160 x 2048), bf16 copy of dt_int
    mamba_wmma_gemm<0, true, MB_DTR, MB_NXP, MB_DI, MB_DI>
        <<<dim3((MB_NXP + 63) / 64, MB_M / 128), blk, 0, stream>>>(
            xab, wb_xp, b_xp, dtBC, dtib);

    // 4) dt = softplus(dt_int @ W_dt + b_dt)    (8192 x 2048 x 128)
    mamba_wmma_gemm<1, false, 0, MB_DI, MB_DTR, MB_DTR>
        <<<dim3(MB_DI / 64, MB_M / 128), blk, 0, stream>>>(
            dtib, wb_dt, b_dt, dt, nullptr);

    // 5) selective scan -> y
    mamba_scan<<<(MB_B * MB_DI) / 256, blk, 0, stream>>>(xa, dt, dtBC, A_log, y);

    // 6) ygb = bf16((y + xa*D) * silu(z))
    mamba_gate<<<(MB_M * MB_DI) / 256, blk, 0, stream>>>(xz, xa, D_ssm, y, ygb);

    // 7) out = yg @ W_out + b_out       (8192 x 1024 x 2048)
    mamba_wmma_gemm<0, false, 0, MB_DIM, MB_DI, MB_DI>
        <<<dim3(MB_DIM / 64, MB_M / 128), blk, 0, stream>>>(
            ygb, wb_out, b_out, out, nullptr);
}
